// DecoderStack_13288628813976
// MI455X (gfx1250) — matrix operand order repaired and checked by reading, not compiled, not run on hardware
//
#include <hip/hip_runtime.h>
#include <hip/hip_bf16.h>

// ---------------------------------------------------------------------------
// Transformer decoder stack for MI455X (gfx1250), bf16 WMMA + f32 accumulate.
// L=6, H=512, NH=8, D=64, F=2048, B=4, T=S=1024. wave32.
// Register-blocked + software-pipelined WMMA kernels:
//  - GEMM wave: 64x32 tile (4x2 frags), double-buffered k-steps
//  - attention wave: 16x64 tile (1x4 frags), double-buffered k-steps
// ---------------------------------------------------------------------------

#define LL 6
#define HH 512
#define NHD 8
#define DD 64
#define FF 2048
#define BB 4
#define TT 1024
#define SS 1024
#define MROWS (BB * TT)   // 4096

typedef __bf16 bf16_t;
typedef __attribute__((ext_vector_type(16))) __bf16 v16bf;
typedef __attribute__((ext_vector_type(8)))  float  v8f;

union Frag32B { v16bf v; uint4 q[2]; };

// ---------------------------------------------------------------------------
// Elementwise converters
// ---------------------------------------------------------------------------
__global__ __launch_bounds__(256) void k_f32_to_bf16(const float* __restrict__ src,
                                                     bf16_t* __restrict__ dst, int n) {
    int i = blockIdx.x * blockDim.x + threadIdx.x;
    if (i < n) dst[i] = (bf16_t)src[i];
}

// W [K,N] f32 row-major  ->  Wt [N,K] bf16 row-major (WMMA B-fragment friendly)
__global__ __launch_bounds__(256) void k_transpose_to_bf16(const float* __restrict__ src,
                                                           bf16_t* __restrict__ dst,
                                                           int K, int N) {
    int i = blockIdx.x * blockDim.x + threadIdx.x;
    if (i >= K * N) return;
    int k = i / N, n = i - k * N;
    dst[(size_t)n * K + k] = (bf16_t)src[i];
}

// V [B,S,H] bf16 -> Vt [B*NH, D, S] bf16  (per-head, feature-major)
__global__ __launch_bounds__(256) void k_make_vt(const bf16_t* __restrict__ V,
                                                 bf16_t* __restrict__ Vt, int Tk) {
    int i = blockIdx.x * blockDim.x + threadIdx.x;
    int total = BB * NHD * DD * Tk;
    if (i >= total) return;
    int s   = i % Tk;
    int r   = i / Tk;
    int dd  = r % DD;
    int bn  = r / DD;
    int b   = bn / NHD, h = bn % NHD;
    Vt[i] = V[((size_t)(b * Tk + s)) * HH + h * DD + dd];
}

// ---------------------------------------------------------------------------
// LayerNorm over rows of 512. One wave32 per row, 16 elements/lane.
// ---------------------------------------------------------------------------
__global__ __launch_bounds__(256) void k_layernorm(const float* __restrict__ x,
                                                   const float* __restrict__ scale,
                                                   const float* __restrict__ bias,
                                                   bf16_t* __restrict__ outB,
                                                   float* __restrict__ outF) {
    int row  = blockIdx.x * 8 + (threadIdx.x >> 5);
    int lane = threadIdx.x & 31;
    const float* rp = x + (size_t)row * HH;
    float4 v[4];
    float sum = 0.f, sq = 0.f;
#pragma unroll
    for (int j = 0; j < 4; ++j) {
        v[j] = *reinterpret_cast<const float4*>(rp + lane * 16 + j * 4);
        sum += v[j].x + v[j].y + v[j].z + v[j].w;
        sq  += v[j].x * v[j].x + v[j].y * v[j].y + v[j].z * v[j].z + v[j].w * v[j].w;
    }
#pragma unroll
    for (int o = 16; o > 0; o >>= 1) {
        sum += __shfl_xor(sum, o);
        sq  += __shfl_xor(sq, o);
    }
    float mu  = sum * (1.0f / HH);
    float var = sq * (1.0f / HH) - mu * mu;
    float rs  = rsqrtf(var + 1e-6f);
#pragma unroll
    for (int j = 0; j < 4; ++j) {
        float e[4] = { v[j].x, v[j].y, v[j].z, v[j].w };
#pragma unroll
        for (int c = 0; c < 4; ++c) {
            int idx = lane * 16 + j * 4 + c;
            float val = (e[c] - mu) * rs * scale[idx] + bias[idx];
            if (outB) outB[(size_t)row * HH + idx] = (bf16_t)val;
            if (outF) outF[(size_t)row * HH + idx] = val;
        }
    }
}

// ---------------------------------------------------------------------------
// Fragment load helpers (A and B: 16-bit WMMA A layout)
// ---------------------------------------------------------------------------
__device__ __forceinline__ void load_a(Frag32B& f, const bf16_t* p) {
    f.q[0] = *reinterpret_cast<const uint4*>(p);
    f.q[1] = *reinterpret_cast<const uint4*>(p + 16);
}
__device__ __forceinline__ void load_b(Frag32B& f, const bf16_t* p) {
    f.q[0] = *reinterpret_cast<const uint4*>(p);
    f.q[1] = *reinterpret_cast<const uint4*>(p + 16);
}

// ---------------------------------------------------------------------------
// Register-blocked, software-pipelined WMMA GEMM:
// C[M,N] = A[M,K](bf16) x Bt[N,K](bf16) (+bias)(+ReLU)(+resid)
// Block = 128 threads (4 waves). Wave tile 64(M) x 32(N) = 4x2 frags,
// k-steps of 32 double-buffered in registers. Requires K % 64 == 0, K >= 128.
// grid = (N/128, M/64)
// ---------------------------------------------------------------------------
__global__ __launch_bounds__(128) void k_gemm_wmma(
        const bf16_t* __restrict__ A, const bf16_t* __restrict__ Bt,
        const float* __restrict__ bias, const float* __restrict__ resid,
        float* __restrict__ outF, bf16_t* __restrict__ outB,
        int M, int N, int K, int relu) {
    int wave = threadIdx.x >> 5;
    int lane = threadIdx.x & 31;
    int mw = blockIdx.y * 64;                 // 4 M-subtiles of 16
    int nw = blockIdx.x * 128 + wave * 32;    // 2 N-subtiles of 16
    if (mw >= M || nw >= N) return;

    v8f acc[4][2] = {};
    const bf16_t* Ar[4];
    const bf16_t* Br[2];
    int hiA = (lane >> 4) * 8;
    int hiB = (lane >> 4) * 8;
#pragma unroll
    for (int i = 0; i < 4; ++i) Ar[i] = A  + (size_t)(mw + 16 * i + (lane & 15)) * K + hiA;
#pragma unroll
    for (int j = 0; j < 2; ++j) Br[j] = Bt + (size_t)(nw + 16 * j + (lane & 15)) * K + hiB;

    Frag32B a0[4], b0[2], a1[4], b1[2];
#pragma unroll
    for (int i = 0; i < 4; ++i) { load_a(a0[i], Ar[i]);      load_a(a1[i], Ar[i] + 32); }
#pragma unroll
    for (int j = 0; j < 2; ++j) { load_b(b0[j], Br[j]);      load_b(b1[j], Br[j] + 32); }

    int k0 = 0;
    for (; k0 < K - 64; k0 += 64) {
#pragma unroll
        for (int i = 0; i < 4; ++i)
#pragma unroll
            for (int j = 0; j < 2; ++j)
                acc[i][j] = __builtin_amdgcn_wmma_f32_16x16x32_bf16(
                    false, a0[i].v, false, b0[j].v, (short)0, acc[i][j], false, false);
#pragma unroll
        for (int i = 0; i < 4; ++i) load_a(a0[i], Ar[i] + k0 + 64);
#pragma unroll
        for (int j = 0; j < 2; ++j) load_b(b0[j], Br[j] + k0 + 64);
#pragma unroll
        for (int i = 0; i < 4; ++i)
#pragma unroll
            for (int j = 0; j < 2; ++j)
                acc[i][j] = __builtin_amdgcn_wmma_f32_16x16x32_bf16(
                    false, a1[i].v, false, b1[j].v, (short)0, acc[i][j], false, false);
#pragma unroll
        for (int i = 0; i < 4; ++i) load_a(a1[i], Ar[i] + k0 + 96);
#pragma unroll
        for (int j = 0; j < 2; ++j) load_b(b1[j], Br[j] + k0 + 96);
    }
#pragma unroll
    for (int i = 0; i < 4; ++i)
#pragma unroll
        for (int j = 0; j < 2; ++j)
            acc[i][j] = __builtin_amdgcn_wmma_f32_16x16x32_bf16(
                false, a0[i].v, false, b0[j].v, (short)0, acc[i][j], false, false);
#pragma unroll
    for (int i = 0; i < 4; ++i)
#pragma unroll
        for (int j = 0; j < 2; ++j)
            acc[i][j] = __builtin_amdgcn_wmma_f32_16x16x32_bf16(
                false, a1[i].v, false, b1[j].v, (short)0, acc[i][j], false, false);

#pragma unroll
    for (int i = 0; i < 4; ++i) {
        int mb = mw + 16 * i + ((lane >> 4) ? 8 : 0);
#pragma unroll
        for (int j = 0; j < 2; ++j) {
            int nc = nw + 16 * j + (lane & 15);
            float bv = bias ? bias[nc] : 0.f;
#pragma unroll
            for (int r = 0; r < 8; ++r) {
                size_t idx = (size_t)(mb + r) * N + nc;
                float val = acc[i][j][r] + bv;
                if (relu) val = fmaxf(val, 0.f);
                if (resid) val += resid[idx];
                if (outF) outF[idx] = val;
                if (outB) outB[idx] = (bf16_t)val;
            }
        }
    }
}

// ---------------------------------------------------------------------------
// Attention logits: logits[bn,q,k] = 0.125 * (Q . K) + bias
// Each wave: 16(q) x 64(k) tile = 1 Q-frag reused over 4 K-frags; K-dim = 64
// is fully unrolled (2 WMMA steps).
// bias_mode 0: bias[q*Tk + k] (causal [T,T]); 1: bias[b*Tk + k] ([B,S]).
// grid = (Tk/256, Tq/16, B*NH), block = 128
// ---------------------------------------------------------------------------
__global__ __launch_bounds__(128) void k_attn_scores(
        const bf16_t* __restrict__ Qb, const bf16_t* __restrict__ Kb,
        const float* __restrict__ bias, int bias_mode,
        float* __restrict__ logits, int Tq, int Tk) {
    int wave = threadIdx.x >> 5;
    int lane = threadIdx.x & 31;
    int kt = blockIdx.x * 256 + wave * 64;    // 4 k-subtiles of 16
    int qt = blockIdx.y * 16;
    int bn = blockIdx.z;
    int b = bn / NHD, h = bn % NHD;

    v8f acc[4] = {};
    int hiA = (lane >> 4) * 8;
    int hiB = (lane >> 4) * 8;
    const bf16_t* Qrow = Qb + (size_t)(b * Tq + qt + (lane & 15)) * HH + h * DD + hiA;
    const bf16_t* Kr[4];
#pragma unroll
    for (int j = 0; j < 4; ++j)
        Kr[j] = Kb + (size_t)(b * Tk + kt + 16 * j + (lane & 15)) * HH + h * DD + hiB;

#pragma unroll
    for (int k0 = 0; k0 < DD; k0 += 32) {
        Frag32B a, bb[4];
        load_a(a, Qrow + k0);
#pragma unroll
        for (int j = 0; j < 4; ++j) load_b(bb[j], Kr[j] + k0);
#pragma unroll
        for (int j = 0; j < 4; ++j)
            acc[j] = __builtin_amdgcn_wmma_f32_16x16x32_bf16(
                false, a.v, false, bb[j].v, (short)0, acc[j], false, false);
    }

    int qb = qt + ((lane >> 4) ? 8 : 0);
#pragma unroll
    for (int j = 0; j < 4; ++j) {
        int kc = kt + 16 * j + (lane & 15);
#pragma unroll
        for (int r = 0; r < 8; ++r) {
            int q = qb + r;
            float bval = (bias_mode == 0) ? bias[(size_t)q * Tk + kc]
                                          : bias[(size_t)b * Tk + kc];
            logits[((size_t)bn * Tq + q) * Tk + kc] = acc[j][r] * 0.125f + bval;
        }
    }
}

// ---------------------------------------------------------------------------
// Softmax over rows of Tk (<=1024). One wave per row; writes bf16 probs
// in place over the start of each f32 row (row stride stays Tk floats).
// ---------------------------------------------------------------------------
__global__ __launch_bounds__(256) void k_softmax(float* __restrict__ logits, int Tk) {
    int row  = blockIdx.x * 8 + (threadIdx.x >> 5);
    int lane = threadIdx.x & 31;
    float* rp = logits + (size_t)row * Tk;
    int cnt = Tk / 32;

    float mx = -3.0e38f;
    float e[32];
    for (int j = 0; j < cnt; ++j) {
        e[j] = rp[lane + j * 32];
        mx = fmaxf(mx, e[j]);
    }
#pragma unroll
    for (int o = 16; o > 0; o >>= 1) mx = fmaxf(mx, __shfl_xor(mx, o));

    float sum = 0.f;
    for (int j = 0; j < cnt; ++j) {
        e[j] = __expf(e[j] - mx);
        sum += e[j];
    }
#pragma unroll
    for (int o = 16; o > 0; o >>= 1) sum += __shfl_xor(sum, o);
    float inv = 1.0f / sum;

    bf16_t* op = reinterpret_cast<bf16_t*>(rp);
    for (int j = 0; j < cnt; ++j) op[lane + j * 32] = (bf16_t)(e[j] * inv);
}

// ---------------------------------------------------------------------------
// Context: ctx[b,q,h,d] = probs[bn,q,:] @ V[bn,:,d]
// probs: bf16 packed at start of each f32 logits row. Vt: [B*NH, D, Tk] bf16.
// Each wave: 16(q) x 64(d) tile = 1 probs-frag + 4 Vt-frags per 32-deep
// k-step, double-buffered. Requires Tk % 64 == 0, Tk >= 128.
// grid = (Tq/16, B*NH), block = 32 (one wave covers all of D=64)
// ---------------------------------------------------------------------------
__global__ __launch_bounds__(32) void k_attn_ctx(
        const float* __restrict__ probs_base, const bf16_t* __restrict__ Vt,
        bf16_t* __restrict__ ctx, int Tq, int Tk) {
    int lane = threadIdx.x & 31;
    int qt = blockIdx.x * 16;
    int bn = blockIdx.y;
    int b = bn / NHD, h = bn % NHD;

    v8f acc[4] = {};
    int hiA = (lane >> 4) * 8;
    int hiB = (lane >> 4) * 8;
    const char* pb = reinterpret_cast<const char*>(probs_base);
    const bf16_t* Arow =
        reinterpret_cast<const bf16_t*>(pb + ((size_t)bn * Tq + qt + (lane & 15)) * Tk * 4) + hiA;
    const bf16_t* Br[4];
#pragma unroll
    for (int j = 0; j < 4; ++j)
        Br[j] = Vt + ((size_t)bn * DD + 16 * j + (lane & 15)) * Tk + hiB;

    Frag32B pa0, pa1, vb0[4], vb1[4];
    load_a(pa0, Arow);
    load_a(pa1, Arow + 32);
#pragma unroll
    for (int j = 0; j < 4; ++j) { load_b(vb0[j], Br[j]); load_b(vb1[j], Br[j] + 32); }

    int k0 = 0;
    for (; k0 < Tk - 64; k0 += 64) {
#pragma unroll
        for (int j = 0; j < 4; ++j)
            acc[j] = __builtin_amdgcn_wmma_f32_16x16x32_bf16(
                false, pa0.v, false, vb0[j].v, (short)0, acc[j], false, false);
        load_a(pa0, Arow + k0 + 64);
#pragma unroll
        for (int j = 0; j < 4; ++j) load_b(vb0[j], Br[j] + k0 + 64);
#pragma unroll
        for (int j = 0; j < 4; ++j)
            acc[j] = __builtin_amdgcn_wmma_f32_16x16x32_bf16(
                false, pa1.v, false, vb1[j].v, (short)0, acc[j], false, false);
        load_a(pa1, Arow + k0 + 96);
#pragma unroll
        for (int j = 0; j < 4; ++j) load_b(vb1[j], Br[j] + k0 + 96);
    }
#pragma unroll
    for (int j = 0; j < 4; ++j)
        acc[j] = __builtin_amdgcn_wmma_f32_16x16x32_bf16(
            false, pa0.v, false, vb0[j].v, (short)0, acc[j], false, false);
#pragma unroll
    for (int j = 0; j < 4; ++j)
        acc[j] = __builtin_amdgcn_wmma_f32_16x16x32_bf16(
            false, pa1.v, false, vb1[j].v, (short)0, acc[j], false, false);

    int qb = qt + ((lane >> 4) ? 8 : 0);
#pragma unroll
    for (int j = 0; j < 4; ++j) {
        int dd = 16 * j + (lane & 15);
#pragma unroll
        for (int r = 0; r < 8; ++r) {
            int q = qb + r;
            ctx[((size_t)(b * Tq + q)) * HH + h * DD + dd] = (bf16_t)acc[j][r];
        }
    }
}

// ---------------------------------------------------------------------------
// Host orchestration
// ---------------------------------------------------------------------------
extern "C" void kernel_launch(void* const* d_in, const int* in_sizes, int n_in,
                              void* d_out, int out_size, void* d_ws, size_t ws_size,
                              hipStream_t stream) {
    const float* dec_in   = (const float*)d_in[0];
    const float* enc_out  = (const float*)d_in[1];
    const float* dec_bias = (const float*)d_in[2];
    const float* enc_bias = (const float*)d_in[3];
    const float* w_attn[8] = { (const float*)d_in[4], (const float*)d_in[5],
                               (const float*)d_in[6], (const float*)d_in[7],
                               (const float*)d_in[8], (const float*)d_in[9],
                               (const float*)d_in[10], (const float*)d_in[11] };
    const float* w1  = (const float*)d_in[12];
    const float* b1  = (const float*)d_in[13];
    const float* w2  = (const float*)d_in[14];
    const float* b2  = (const float*)d_in[15];
    const float* ln1s = (const float*)d_in[16];
    const float* ln1b = (const float*)d_in[17];
    const float* ln2s = (const float*)d_in[18];
    const float* ln2b = (const float*)d_in[19];
    const float* ln3s = (const float*)d_in[20];
    const float* ln3b = (const float*)d_in[21];
    const float* lnfs = (const float*)d_in[22];
    const float* lnfb = (const float*)d_in[23];

    // ---- workspace carve ----
    char* ws = (char*)d_ws;
    size_t off = 0;
    auto carve = [&](size_t bytes) -> char* {
        char* p = ws + off;
        off = (off + bytes + 255) & ~(size_t)255;
        return p;
    };
    bf16_t* wt_attn = (bf16_t*)carve((size_t)8 * LL * HH * HH * 2);   // transposed bf16 QKVO (self+cross)
    bf16_t* wt_w1   = (bf16_t*)carve((size_t)LL * HH * FF * 2);       // [F,H] per layer
    bf16_t* wt_w2   = (bf16_t*)carve((size_t)LL * FF * HH * 2);       // [H,F] per layer
    float*  xbuf    = (float*) carve((size_t)MROWS * HH * 4);         // residual state
    bf16_t* enc_bf  = (bf16_t*)carve((size_t)BB * SS * HH * 2);
    bf16_t* y_bf    = (bf16_t*)carve((size_t)MROWS * HH * 2);
    bf16_t* q_bf    = (bf16_t*)carve((size_t)MROWS * HH * 2);
    bf16_t* k_bf    = (bf16_t*)carve((size_t)BB * SS * HH * 2);
    bf16_t* v_bf    = (bf16_t*)carve((size_t)BB * SS * HH * 2);
    bf16_t* vt_bf   = (bf16_t*)carve((size_t)BB * NHD * DD * SS * 2);
    bf16_t* ctx_bf  = (bf16_t*)carve((size_t)MROWS * HH * 2);
    bf16_t* h_bf    = (bf16_t*)carve((size_t)MROWS * FF * 2);
    float*  logits  = (float*) carve((size_t)BB * NHD * TT * SS * 4); // L2-resident (134MB < 192MB)

    // ---- one-time conversions (per call; deterministic) ----
    for (int t = 0; t < 8; ++t)
        for (int l = 0; l < LL; ++l)
            k_transpose_to_bf16<<<(HH * HH + 255) / 256, 256, 0, stream>>>(
                w_attn[t] + (size_t)l * HH * HH,
                wt_attn + ((size_t)t * LL + l) * HH * HH, HH, HH);
    for (int l = 0; l < LL; ++l)
        k_transpose_to_bf16<<<(HH * FF + 255) / 256, 256, 0, stream>>>(
            w1 + (size_t)l * HH * FF, wt_w1 + (size_t)l * HH * FF, HH, FF);
    for (int l = 0; l < LL; ++l)
        k_transpose_to_bf16<<<(FF * HH + 255) / 256, 256, 0, stream>>>(
            w2 + (size_t)l * FF * HH, wt_w2 + (size_t)l * FF * HH, FF, HH);

    hipMemcpyAsync(xbuf, dec_in, (size_t)MROWS * HH * 4, hipMemcpyDeviceToDevice, stream);
    k_f32_to_bf16<<<(BB * SS * HH + 255) / 256, 256, 0, stream>>>(enc_out, enc_bf, BB * SS * HH);

    auto gemm = [&](const bf16_t* A, const bf16_t* Bt, const float* bias, const float* resid,
                    float* oF, bf16_t* oB, int M, int N, int K, int relu) {
        k_gemm_wmma<<<dim3(N / 128, M / 64), 128, 0, stream>>>(A, Bt, bias, resid, oF, oB,
                                                               M, N, K, relu);
    };

    for (int l = 0; l < LL; ++l) {
        const size_t wA = (size_t)HH * HH;
        const bf16_t* swq = wt_attn + ((size_t)0 * LL + l) * wA;
        const bf16_t* swk = wt_attn + ((size_t)1 * LL + l) * wA;
        const bf16_t* swv = wt_attn + ((size_t)2 * LL + l) * wA;
        const bf16_t* swo = wt_attn + ((size_t)3 * LL + l) * wA;
        const bf16_t* cwq = wt_attn + ((size_t)4 * LL + l) * wA;
        const bf16_t* cwk = wt_attn + ((size_t)5 * LL + l) * wA;
        const bf16_t* cwv = wt_attn + ((size_t)6 * LL + l) * wA;
        const bf16_t* cwo = wt_attn + ((size_t)7 * LL + l) * wA;

        // ---- 1. self-attention ----
        k_layernorm<<<MROWS / 8, 256, 0, stream>>>(xbuf, ln1s + l * HH, ln1b + l * HH,
                                                   y_bf, nullptr);
        gemm(y_bf, swq, nullptr, nullptr, nullptr, q_bf, MROWS, HH, HH, 0);
        gemm(y_bf, swk, nullptr, nullptr, nullptr, k_bf, MROWS, HH, HH, 0);
        gemm(y_bf, swv, nullptr, nullptr, nullptr, v_bf, MROWS, HH, HH, 0);
        k_attn_scores<<<dim3(TT / 256, TT / 16, BB * NHD), 128, 0, stream>>>(
            q_bf, k_bf, dec_bias, 0, logits, TT, TT);
        k_softmax<<<(BB * NHD * TT) / 8, 256, 0, stream>>>(logits, TT);
        k_make_vt<<<(BB * NHD * DD * TT) / 256, 256, 0, stream>>>(v_bf, vt_bf, TT);
        k_attn_ctx<<<dim3(TT / 16, BB * NHD), 32, 0, stream>>>(logits, vt_bf, ctx_bf, TT, TT);
        gemm(ctx_bf, swo, nullptr, xbuf, xbuf, nullptr, MROWS, HH, HH, 0);

        // ---- 2. cross-attention ----
        k_layernorm<<<MROWS / 8, 256, 0, stream>>>(xbuf, ln2s + l * HH, ln2b + l * HH,
                                                   y_bf, nullptr);
        gemm(y_bf,   cwq, nullptr, nullptr, nullptr, q_bf, MROWS, HH, HH, 0);
        gemm(enc_bf, cwk, nullptr, nullptr, nullptr, k_bf, BB * SS, HH, HH, 0);
        gemm(enc_bf, cwv, nullptr, nullptr, nullptr, v_bf, BB * SS, HH, HH, 0);
        k_attn_scores<<<dim3(SS / 256, TT / 16, BB * NHD), 128, 0, stream>>>(
            q_bf, k_bf, enc_bias, 1, logits, TT, SS);
        k_softmax<<<(BB * NHD * TT) / 8, 256, 0, stream>>>(logits, SS);
        k_make_vt<<<(BB * NHD * DD * SS) / 256, 256, 0, stream>>>(v_bf, vt_bf, SS);
        k_attn_ctx<<<dim3(TT / 16, BB * NHD), 32, 0, stream>>>(logits, vt_bf, ctx_bf, TT, SS);
        gemm(ctx_bf, cwo, nullptr, xbuf, xbuf, nullptr, MROWS, HH, HH, 0);

        // ---- 3. FFN ----
        k_layernorm<<<MROWS / 8, 256, 0, stream>>>(xbuf, ln3s + l * HH, ln3b + l * HH,
                                                   y_bf, nullptr);
        gemm(y_bf, wt_w1 + (size_t)l * HH * FF, b1 + l * FF, nullptr,
             nullptr, h_bf, MROWS, FF, HH, 1);
        gemm(h_bf, wt_w2 + (size_t)l * FF * HH, b2 + l * HH, xbuf,
             xbuf, nullptr, MROWS, HH, FF, 0);
    }

    // ---- final LayerNorm -> d_out (f32) ----
    k_layernorm<<<MROWS / 8, 256, 0, stream>>>(xbuf, lnfs, lnfb, nullptr, (float*)d_out);
}
